// Experts_54425825575561
// MI455X (gfx1250) — matrix operand order repaired and checked by reading, not compiled, not run on hardware
//
#include <hip/hip_runtime.h>
#include <math.h>

typedef unsigned short ushort_t;
typedef __attribute__((ext_vector_type(16))) __bf16 v16bf;
typedef __attribute__((ext_vector_type(8)))  float  v8f;

#define T_TOK 4096
#define D_DIM 2048
#define H_DIM 8192
#define R_RANK 16
#define SCALING 2.0f

#define BM 128
#define BN 128
#define BK 32
#define LDSTR 48   // bf16 elems per LDS row (32 data + 16 pad, keeps 16B alignment)

#if defined(__gfx1250__) && __has_builtin(__builtin_amdgcn_global_load_async_to_lds_b128)
#define USE_ASYNC_LDS 1
#else
#define USE_ASYNC_LDS 0
#endif

__device__ __forceinline__ ushort_t f2bf(float f) {
  unsigned int u = __float_as_uint(f);
  unsigned int r = (u + 0x7FFFu + ((u >> 16) & 1u)) >> 16;  // RNE
  return (ushort_t)r;
}
__device__ __forceinline__ float bf2f(ushort_t h) {
  return __uint_as_float(((unsigned int)h) << 16);
}
__device__ __forceinline__ float gelu_new_f(float x) {
  const float c0 = 0.7978845608028654f;  // sqrt(2/pi)
  return 0.5f * x * (1.0f + tanhf(c0 * (x + 0.044715f * x * x * x)));
}
__device__ __forceinline__ void store_out(float* p, float v) { *p = v; }
__device__ __forceinline__ void store_out(ushort_t* p, float v) { *p = f2bf(v); }

#if USE_ASYNC_LDS
// Builtin signature (from hipcc diagnostic): param0 = int4 AS(1)* (global src),
// param1 = int4 AS(3)* (LDS dst), then imm offset, imm cpol.
typedef int v4i32 __attribute__((vector_size(16)));
typedef __attribute__((address_space(1))) v4i32* gbl_v4i_ptr;
typedef __attribute__((address_space(3))) v4i32* lds_v4i_ptr;

// one lane moves 16 bytes global -> LDS, tracked by ASYNCcnt (no VGPR data path)
__device__ __forceinline__ void async_cp16(const ushort_t* g, ushort_t* l) {
  __builtin_amdgcn_global_load_async_to_lds_b128(
      (gbl_v4i_ptr)(g), (lds_v4i_ptr)(l), 0, 0);
}
__device__ __forceinline__ void wait_async0() {
#if __has_builtin(__builtin_amdgcn_s_wait_asynccnt)
  __builtin_amdgcn_s_wait_asynccnt(0);
#else
  asm volatile("s_wait_asynccnt 0" ::: "memory");
#endif
}
#endif

// ---------------- cast f32 -> bf16, vectorized ----------------
__global__ void cast_f32_to_bf16(const float* __restrict__ src,
                                 ushort_t* __restrict__ dst, int n4) {
  int i = blockIdx.x * blockDim.x + threadIdx.x;
  if (i >= n4) return;
  float4 v = ((const float4*)src)[i];
  ushort4 o;
  o.x = f2bf(v.x); o.y = f2bf(v.y); o.z = f2bf(v.z); o.w = f2bf(v.w);
  ((ushort4*)dst)[i] = o;
}

// ---------------- z[t,r] = SCALING * sum_d X[t,d] * A[e_t,r,d] ----------------
template <typename InT>
__global__ void lora_z_kernel(const InT* __restrict__ X,      // [T,K]
                              const float* __restrict__ A,    // [E,R,K]
                              const int* __restrict__ eidx,   // [T]
                              float* __restrict__ Z,          // [T,R]
                              int K) {
  extern __shared__ float sx[];
  const int t = blockIdx.x;
  const InT* xr = X + (size_t)t * K;
  for (int i = threadIdx.x; i < K; i += blockDim.x) {
    float v;
    if constexpr (sizeof(InT) == 2) v = bf2f((ushort_t)xr[i]);
    else                            v = (float)xr[i];
    sx[i] = v;
  }
  __syncthreads();
  const int e = eidx[t];
  const int r = threadIdx.x >> 4;   // 0..15 (256 threads)
  const int j = threadIdx.x & 15;
  const float* ar = A + ((size_t)e * R_RANK + r) * K;
  float s = 0.f;
  for (int d = j; d < K; d += 16) s += sx[d] * ar[d];
#pragma unroll
  for (int m = 8; m >= 1; m >>= 1) s += __shfl_xor(s, m, 32);  // stays in 16-group
  if (j == 0) Z[(size_t)t * R_RANK + r] = SCALING * s;
}

// ---------------- bf16 WMMA GEMM + bias + rank-16 LoRA epilogue ----------------
// Out[t,n] = sum_k Xb[t,k]*Wb[n,k] + bias[n] + dot(Z[t,:], Bl[e_t,n,:])  (+gelu)
template <bool GELU, typename OutT>
__global__ void __launch_bounds__(256)
gemm_wmma_lora(const ushort_t* __restrict__ Xb,   // [T,K] bf16
               const ushort_t* __restrict__ Wb,   // [N,K] bf16
               const float* __restrict__ bias,    // [N]
               const float* __restrict__ Z,       // [T,16] pre-scaled
               const float* __restrict__ Bl,      // [E,N,16]
               const int* __restrict__ eidx,      // [T]
               OutT* __restrict__ Out,            // [T,N]
               int T, int N, int K) {
  (void)T;
  __shared__ __align__(16) ushort_t sA[2][BM * LDSTR];
  __shared__ __align__(16) ushort_t sB[2][BN * LDSTR];

  const int tid  = threadIdx.x;
  const int lane = tid & 31;        // wave32
  const int wave = tid >> 5;        // 8 waves
  const int l15  = lane & 15;
  const int half = lane >> 4;
  const int waveM = (wave & 1) * 64;   // 2x4 wave grid -> 64x32 per wave
  const int waveN = (wave >> 1) * 32;

  const int blockM = blockIdx.y * BM;
  const int blockN = blockIdx.x * BN;

  // each thread stages 16 bf16 (32B) of A and of B per K-step
  const int ldRow = tid >> 1;           // 0..127
  const int ldK   = (tid & 1) * 16;     // 0 or 16
  const ushort_t* gA = Xb + (size_t)(blockM + ldRow) * K + ldK;
  const ushort_t* gB = Wb + (size_t)(blockN + ldRow) * K + ldK;
  ushort_t* lA0 = &sA[0][ldRow * LDSTR + ldK];
  ushort_t* lB0 = &sB[0][ldRow * LDSTR + ldK];

  const int nK = K / BK;
  int buf = 0;

#if USE_ASYNC_LDS
  // stage K-tile 0 via async LDS DMA
  async_cp16(gA,     lA0);
  async_cp16(gA + 8, lA0 + 8);
  async_cp16(gB,     lB0);
  async_cp16(gB + 8, lB0 + 8);
  wait_async0();
  __syncthreads();
#endif

  v8f acc[4][2];
#pragma unroll
  for (int mi = 0; mi < 4; ++mi)
#pragma unroll
    for (int ni = 0; ni < 2; ++ni)
      acc[mi][ni] = (v8f){0.f, 0.f, 0.f, 0.f, 0.f, 0.f, 0.f, 0.f};

  union Frag { v16bf v; uint4 u[2]; };

  for (int kt = 0; kt < nK; ++kt) {
    const bool more = (kt + 1) < nK;
#if USE_ASYNC_LDS
    if (more) {  // kick async copies for next tile into the other buffer
      const ushort_t* nA = gA + (size_t)(kt + 1) * BK;
      const ushort_t* nB = gB + (size_t)(kt + 1) * BK;
      ushort_t* dA = lA0 + (buf ^ 1) * (BM * LDSTR);
      ushort_t* dB = lB0 + (buf ^ 1) * (BN * LDSTR);
      async_cp16(nA,     dA);
      async_cp16(nA + 8, dA + 8);
      async_cp16(nB,     dB);
      async_cp16(nB + 8, dB + 8);
    }
#else
    {  // synchronous fallback: regs -> LDS (single buffer)
      const ushort_t* nA = gA + (size_t)kt * BK;
      const ushort_t* nB = gB + (size_t)kt * BK;
      uint4 a0 = ((const uint4*)nA)[0];
      uint4 a1 = ((const uint4*)nA)[1];
      uint4 b0 = ((const uint4*)nB)[0];
      uint4 b1 = ((const uint4*)nB)[1];
      if (kt) __syncthreads();
      *(uint4*)(lA0)     = a0;
      *(uint4*)(lA0 + 8) = a1;
      *(uint4*)(lB0)     = b0;
      *(uint4*)(lB0 + 8) = b1;
      __syncthreads();
    }
#endif

    // A fragment (16x32 bf16, ISA layout): lanes 0-15 hold K[0..7],K[16..23];
    // lanes 16-31 hold K[8..15],K[24..31] for row M=l15.
    Frag a[4], b[2];
#pragma unroll
    for (int mi = 0; mi < 4; ++mi) {
      const ushort_t* p = &sA[buf][(waveM + mi * 16 + l15) * LDSTR];
      a[mi].u[0] = *(const uint4*)(p + 8 * half);
      a[mi].u[1] = *(const uint4*)(p + 16 + 8 * half);
    }
    // B fragment (32x16 bf16): lanes 0-15 hold K[0..7],K[16..23]; lanes 16-31
    // hold K[8..15],K[24..31] for column N=l15.
#pragma unroll
    for (int ni = 0; ni < 2; ++ni) {
      const ushort_t* p = &sB[buf][(waveN + ni * 16 + l15) * LDSTR + 8 * half];
      b[ni].u[0] = *(const uint4*)(p);
      b[ni].u[1] = *(const uint4*)(p + 16);
    }

#pragma unroll
    for (int mi = 0; mi < 4; ++mi)
#pragma unroll
      for (int ni = 0; ni < 2; ++ni)
        acc[mi][ni] = __builtin_amdgcn_wmma_f32_16x16x32_bf16(
            false, a[mi].v, false, b[ni].v, (short)0, acc[mi][ni], false, false);

#if USE_ASYNC_LDS
    if (more) {
      wait_async0();      // this wave's copies into buf^1 are done
      __syncthreads();    // everyone's copies are done; buf readers finished
      buf ^= 1;
    }
#endif
  }

  // ---- epilogue: bias + rank-16 LoRA + (gelu) + store ----
  // C layout: v8f elem i -> row (i + 8*half), col l15 within a 16x16 tile.
  int   cols[2];
  float bc[2];
#pragma unroll
  for (int ni = 0; ni < 2; ++ni) {
    cols[ni] = blockN + waveN + ni * 16 + l15;
    bc[ni]   = bias[cols[ni]];
  }
#pragma unroll
  for (int mi = 0; mi < 4; ++mi) {
    const int row0 = blockM + waveM + mi * 16 + 8 * half;
#pragma unroll
    for (int i = 0; i < 8; ++i) {
      const int t = row0 + i;
      const int e = eidx[t];
      const float4* zr = (const float4*)(Z + (size_t)t * R_RANK);
      float4 z0 = zr[0], z1 = zr[1], z2 = zr[2], z3 = zr[3];
#pragma unroll
      for (int ni = 0; ni < 2; ++ni) {
        const float4* br = (const float4*)(Bl + ((size_t)e * N + cols[ni]) * R_RANK);
        float4 b0 = br[0], b1 = br[1], b2 = br[2], b3 = br[3];
        float s = z0.x * b0.x + z0.y * b0.y + z0.z * b0.z + z0.w * b0.w;
        s += z1.x * b1.x + z1.y * b1.y + z1.z * b1.z + z1.w * b1.w;
        s += z2.x * b2.x + z2.y * b2.y + z2.z * b2.z + z2.w * b2.w;
        s += z3.x * b3.x + z3.y * b3.y + z3.z * b3.z + z3.w * b3.w;
        float v = acc[mi][ni][i] + bc[ni] + s;
        if constexpr (GELU) v = gelu_new_f(v);
        store_out(Out + (size_t)t * N + cols[ni], v);
      }
    }
  }
}

extern "C" void kernel_launch(void* const* d_in, const int* in_sizes, int n_in,
                              void* d_out, int out_size, void* d_ws, size_t ws_size,
                              hipStream_t stream) {
  (void)in_sizes; (void)n_in; (void)out_size; (void)ws_size;
  const float* x   = (const float*)d_in[0];
  const int*   eid = (const int*)d_in[1];
  const float* W1  = (const float*)d_in[2];
  const float* b1  = (const float*)d_in[3];
  const float* A1  = (const float*)d_in[4];
  const float* B1  = (const float*)d_in[5];
  const float* W2  = (const float*)d_in[6];
  const float* b2  = (const float*)d_in[7];
  const float* A2  = (const float*)d_in[8];
  const float* B2  = (const float*)d_in[9];
  float* out = (float*)d_out;

  // workspace layout (bytes)
  char* w = (char*)d_ws;
  ushort_t* xb  = (ushort_t*)(w);                          // 16 MB  [T,D] bf16
  ushort_t* w1b = (ushort_t*)(w + (size_t)16777216);       // 32 MB  [H,D] bf16
  ushort_t* w2b = (ushort_t*)(w + (size_t)50331648);       // 32 MB  [D,H] bf16
  ushort_t* yb  = (ushort_t*)(w + (size_t)83886080);       // 64 MB  [T,H] bf16
  float*    z1  = (float*)(w + (size_t)150994944);         // 256 KB [T,16]
  float*    z2  = (float*)(w + (size_t)151257088);         // 256 KB [T,16]

  // 1) casts to bf16
  {
    int n4 = (T_TOK * D_DIM) / 4;
    cast_f32_to_bf16<<<n4 / 256, 256, 0, stream>>>(x, xb, n4);
    n4 = (H_DIM * D_DIM) / 4;
    cast_f32_to_bf16<<<n4 / 256, 256, 0, stream>>>(W1, w1b, n4);
    cast_f32_to_bf16<<<n4 / 256, 256, 0, stream>>>(W2, w2b, n4);
  }
  // 2) z1 = 2 * x @ A1[e]^T   (f32 inputs)
  lora_z_kernel<float><<<T_TOK, 256, D_DIM * 4, stream>>>(x, A1, eid, z1, D_DIM);
  // 3) fc1: y = gelu(x@W1^T + b1 + z1@B1[e]^T), stored bf16
  dim3 g1(H_DIM / BN, T_TOK / BM);
  gemm_wmma_lora<true, ushort_t><<<g1, 256, 0, stream>>>(
      xb, w1b, b1, z1, B1, eid, yb, T_TOK, H_DIM, D_DIM);
  // 4) z2 = 2 * y @ A2[e]^T   (bf16 y)
  lora_z_kernel<ushort_t><<<T_TOK, 256, H_DIM * 4, stream>>>(yb, A2, eid, z2, H_DIM);
  // 5) fc2: out = y@W2^T + b2 + z2@B2[e]^T, f32 output
  dim3 g2(D_DIM / BN, T_TOK / BM);
  gemm_wmma_lora<false, float><<<g2, 256, 0, stream>>>(
      yb, w2b, b2, z2, B2, eid, out, T_TOK, D_DIM, H_DIM);
}
